// SRNN2_20091857011144
// MI455X (gfx1250) — matrix operand order repaired and checked by reading, not compiled, not run on hardware
//
#include <hip/hip_runtime.h>

typedef __attribute__((ext_vector_type(16))) _Float16 v16h;
typedef __attribute__((ext_vector_type(8)))  float    v8f;

// ---- branchless gate math on CDNA5 hardware transcendentals ----
__device__ __forceinline__ float fast_exp(float x) {
#if __has_builtin(__builtin_amdgcn_exp2f)
  return __builtin_amdgcn_exp2f(x * 1.44269504088896340736f);  // v_exp_f32
#else
  return __expf(x);
#endif
}

__device__ __forceinline__ float fast_rcp(float x) {
#if __has_builtin(__builtin_amdgcn_rcpf)
  return __builtin_amdgcn_rcpf(x);  // v_rcp_f32
#else
  return 1.0f / x;
#endif
}

__device__ __forceinline__ float fast_sigmoid(float x) {
  return fast_rcp(1.0f + fast_exp(-x));
}

__device__ __forceinline__ float fast_tanh(float x) {
#if __has_builtin(__builtin_amdgcn_tanhf)
  return __builtin_amdgcn_tanhf(x);  // v_tanh_f32 (gfx1250 TRANS)
#else
  return 2.0f * fast_sigmoid(2.0f * x) - 1.0f;  // branchless fallback
#endif
}

__device__ __forceinline__ v8f wmma_f16(v16h a, v16h b, v8f c) {
  // D(f32 16x16) = A(f16 16x32) * B(f16 32x16) + C
  return __builtin_amdgcn_wmma_f32_16x16x32_f16(false, a, false, b, (short)0, c,
                                                false, false);
}

// ---- operand gathers matching the CDNA5 WMMA VGPR layouts (wave32) ----
// A (16x32, f16): lane(llo,lhi) holds row M=llo, K = kbase + {8*lhi..8*lhi+7}
// and kbase + {16+8*lhi .. 23+8*lhi}  (ISA 7.12.2, 16-bit A layout).
__device__ __forceinline__ v16h load_a_f32(const float* __restrict__ row,
                                           int kbase, int lhi) {
  v16h a;
  const int k0 = kbase + 8 * lhi;
#pragma unroll
  for (int i = 0; i < 8; ++i) a[i] = (_Float16)row[k0 + i];
#pragma unroll
  for (int i = 0; i < 8; ++i) a[8 + i] = (_Float16)row[k0 + 16 + i];
  return a;
}

__device__ __forceinline__ v16h load_a_f16(const _Float16* __restrict__ row,
                                           int kbase, int lhi) {
  v16h a;
  const int k0 = kbase + 8 * lhi;
#pragma unroll
  for (int i = 0; i < 8; ++i) a[i] = row[k0 + i];
#pragma unroll
  for (int i = 0; i < 8; ++i) a[8 + i] = row[k0 + 16 + i];
  return a;
}

// B (32x16, f16): lane holds column N=llo, K = kbase + 8*lhi + i (i<8), kbase + 16 + 8*lhi + (i-8) (i>=8).
// Source is a row-major f32 weight [Ktot x ld]; converted to f16 on gather.
__device__ __forceinline__ v16h load_b_f32(const float* __restrict__ W, int ld,
                                           int kbase, int col, int lhi) {
  v16h b;
#pragma unroll
  for (int i = 0; i < 16; ++i)
    b[i] = (_Float16)W[(size_t)(kbase + 8 * lhi + (i < 8 ? i : i + 8)) * ld + col];
  return b;
}

__device__ __forceinline__ v16h load_b_f32_pad(const float* __restrict__ W,
                                               int ld, int kbase, int col,
                                               int lhi, int ncols) {
  v16h b;
#pragma unroll
  for (int i = 0; i < 16; ++i)
    b[i] = (col < ncols)
               ? (_Float16)W[(size_t)(kbase + 8 * lhi + (i < 8 ? i : i + 8)) * ld + col]
               : (_Float16)0.0f;
  return b;
}

// =====================  Layer 0  =====================
// 16384 sequences, 32 steps, D=32 -> H=64. One wave32 per 16-row tile.
// x: [2048, 256, 32] f32.  Sequence n = brick*256 + b reads x[brick*32+s, b, :].
// Final h written f16 to last0[n, 0:64]  (== layer-1 input [64][256][64]).
__global__ __launch_bounds__(32) void fastgrnn_l0(
    const float* __restrict__ x, const float* __restrict__ W0,
    const float* __restrict__ U0, const float* __restrict__ bg,
    const float* __restrict__ bu, const float* __restrict__ zeta,
    const float* __restrict__ nu, _Float16* __restrict__ last0) {
  __shared__ __align__(16) _Float16 L[16][72];  // h staging tile (padded rows)
  const int lane = threadIdx.x & 31;
  const int llo = lane & 15, lhi = lane >> 4;
  const int n0 = blockIdx.x * 16;  // 16 consecutive sequences
  const int brick = n0 >> 8;
  const int bIdx = n0 & 255;

  // Weights resident in registers, B layout, f16.
  v16h W0B[4];     // D=32 -> single k-chunk
  v16h U0B[2][4];  // H=64 -> two k-chunks
#pragma unroll
  for (int j = 0; j < 4; ++j) {
    W0B[j] = load_b_f32(W0, 64, 0, 16 * j + llo, lhi);
#pragma unroll
    for (int kc = 0; kc < 2; ++kc)
      U0B[kc][j] = load_b_f32(U0, 64, 32 * kc, 16 * j + llo, lhi);
  }

  float bgv[4], buv[4];
#pragma unroll
  for (int j = 0; j < 4; ++j) {
    bgv[j] = bg[16 * j + llo];
    buv[j] = bu[16 * j + llo];
  }
  const float zs = fast_sigmoid(zeta[0]);
  const float ns = fast_sigmoid(nu[0]);

  // h in C/D layout: h[j][v] = element (row v+8*lhi, col 16*j+llo)
  float h[4][8];
#pragma unroll
  for (int j = 0; j < 4; ++j)
#pragma unroll
    for (int v = 0; v < 8; ++v) h[j][v] = 0.0f;

  for (int s = 0; s < 32; ++s) {
    const float* xrow = x + (size_t)((brick * 32 + s) * 256 + bIdx + llo) * 32;
    if (s < 31)
      __builtin_prefetch(
          x + (size_t)((brick * 32 + s + 1) * 256 + bIdx + llo) * 32, 0, 0);
    v16h ax = load_a_f32(xrow, 0, lhi);

    // h (C layout, f32) -> LDS f16 tile -> A layout (two k-chunks)
#pragma unroll
    for (int j = 0; j < 4; ++j)
#pragma unroll
      for (int v = 0; v < 8; ++v)
        L[v + 8 * lhi][16 * j + llo] = (_Float16)h[j][v];
    v16h ah0 = load_a_f16(&L[llo][0], 0, lhi);
    v16h ah1 = load_a_f16(&L[llo][0], 32, lhi);

#pragma unroll
    for (int j = 0; j < 4; ++j) {
      v8f acc = {};
      acc = wmma_f16(ax, W0B[j], acc);      // x_t @ W0
      acc = wmma_f16(ah0, U0B[0][j], acc);  // h @ U0 (k 0:32)
      acc = wmma_f16(ah1, U0B[1][j], acc);  // h @ U0 (k 32:64)
#pragma unroll
      for (int v = 0; v < 8; ++v) {
        const float pre = acc[v];
        const float z = fast_sigmoid(pre + bgv[j]);
        const float c = fast_tanh(pre + buv[j]);
        h[j][v] = z * h[j][v] + fmaf(zs, 1.0f - z, ns) * c;
      }
    }
  }

  // Epilogue: stage final h in LDS, then write vectorized 64B row-halves.
#pragma unroll
  for (int j = 0; j < 4; ++j)
#pragma unroll
    for (int v = 0; v < 8; ++v)
      L[v + 8 * lhi][16 * j + llo] = (_Float16)h[j][v];
  {
    const int row = lane >> 1;            // 0..15
    const int cpart = (lane & 1) * 32;    // 0 or 32 (halfs)
    const int4* s4 = (const int4*)&L[row][cpart];
    int4* d4 = (int4*)(last0 + (size_t)(n0 + row) * 64 + cpart);
    d4[0] = s4[0];
    d4[1] = s4[1];
    d4[2] = s4[2];
    d4[3] = s4[3];
  }
}

// =====================  Layer 1 + output head  =====================
// 256 rows, 64 steps, H0=64 -> H1=64; then h @ Wout + Bout -> [256,35].
// One wave32 per 16-row tile (16 blocks).
__global__ __launch_bounds__(32) void fastgrnn_l1(
    const _Float16* __restrict__ xs,  // [64][256][64] f16 (layer-0 output)
    const float* __restrict__ W1, const float* __restrict__ U1,
    const float* __restrict__ bg, const float* __restrict__ bu,
    const float* __restrict__ zeta, const float* __restrict__ nu,
    const float* __restrict__ Wout, const float* __restrict__ Bout,
    float* __restrict__ out) {
  __shared__ __align__(16) _Float16 L[16][72];
  const int lane = threadIdx.x & 31;
  const int llo = lane & 15, lhi = lane >> 4;
  const int r0 = blockIdx.x * 16;  // batch rows

  v16h W1B[2][4], U1B[2][4];
#pragma unroll
  for (int kc = 0; kc < 2; ++kc)
#pragma unroll
    for (int j = 0; j < 4; ++j) {
      W1B[kc][j] = load_b_f32(W1, 64, 32 * kc, 16 * j + llo, lhi);
      U1B[kc][j] = load_b_f32(U1, 64, 32 * kc, 16 * j + llo, lhi);
    }

  float bgv[4], buv[4];
#pragma unroll
  for (int j = 0; j < 4; ++j) {
    bgv[j] = bg[16 * j + llo];
    buv[j] = bu[16 * j + llo];
  }
  const float zs = fast_sigmoid(zeta[0]);
  const float ns = fast_sigmoid(nu[0]);

  float h[4][8];
#pragma unroll
  for (int j = 0; j < 4; ++j)
#pragma unroll
    for (int v = 0; v < 8; ++v) h[j][v] = 0.0f;

  for (int s = 0; s < 64; ++s) {
    const _Float16* xrow = xs + (size_t)(s * 256 + r0 + llo) * 64;
    if (s < 63)
      __builtin_prefetch(xs + (size_t)((s + 1) * 256 + r0 + llo) * 64, 0, 0);
    v16h ax0 = load_a_f16(xrow, 0, lhi);
    v16h ax1 = load_a_f16(xrow, 32, lhi);

#pragma unroll
    for (int j = 0; j < 4; ++j)
#pragma unroll
      for (int v = 0; v < 8; ++v)
        L[v + 8 * lhi][16 * j + llo] = (_Float16)h[j][v];
    v16h ah0 = load_a_f16(&L[llo][0], 0, lhi);
    v16h ah1 = load_a_f16(&L[llo][0], 32, lhi);

#pragma unroll
    for (int j = 0; j < 4; ++j) {
      v8f acc = {};
      acc = wmma_f16(ax0, W1B[0][j], acc);
      acc = wmma_f16(ax1, W1B[1][j], acc);
      acc = wmma_f16(ah0, U1B[0][j], acc);
      acc = wmma_f16(ah1, U1B[1][j], acc);
#pragma unroll
      for (int v = 0; v < 8; ++v) {
        const float pre = acc[v];
        const float z = fast_sigmoid(pre + bgv[j]);
        const float c = fast_tanh(pre + buv[j]);
        h[j][v] = z * h[j][v] + fmaf(zs, 1.0f - z, ns) * c;
      }
    }
  }

  // Output head: h_final @ Wout + Bout   (Wout zero-padded 35 -> 48 cols)
  v16h WoB[2][3];
#pragma unroll
  for (int kc = 0; kc < 2; ++kc)
#pragma unroll
    for (int j = 0; j < 3; ++j)
      WoB[kc][j] = load_b_f32_pad(Wout, 35, 32 * kc, 16 * j + llo, lhi, 35);

#pragma unroll
  for (int j = 0; j < 4; ++j)
#pragma unroll
    for (int v = 0; v < 8; ++v)
      L[v + 8 * lhi][16 * j + llo] = (_Float16)h[j][v];
  v16h ah0 = load_a_f16(&L[llo][0], 0, lhi);
  v16h ah1 = load_a_f16(&L[llo][0], 32, lhi);

#pragma unroll
  for (int j = 0; j < 3; ++j) {
    v8f acc = {};
    acc = wmma_f16(ah0, WoB[0][j], acc);
    acc = wmma_f16(ah1, WoB[1][j], acc);
    const int col = 16 * j + llo;
    if (col < 35) {
      const float bo = Bout[col];
#pragma unroll
      for (int v = 0; v < 8; ++v)
        out[(size_t)(r0 + v + 8 * lhi) * 35 + col] = acc[v] + bo;
    }
  }
}

extern "C" void kernel_launch(void* const* d_in, const int* in_sizes, int n_in,
                              void* d_out, int out_size, void* d_ws,
                              size_t ws_size, hipStream_t stream) {
  (void)in_sizes; (void)n_in; (void)out_size; (void)ws_size;
  const float* x    = (const float*)d_in[0];
  // d_in[1] = brickSize (int scalar) == 32, baked in.
  const float* W0   = (const float*)d_in[2];
  const float* U0   = (const float*)d_in[3];
  const float* bg0  = (const float*)d_in[4];
  const float* bu0  = (const float*)d_in[5];
  const float* z0   = (const float*)d_in[6];
  const float* nu0  = (const float*)d_in[7];
  const float* W1   = (const float*)d_in[8];
  const float* U1   = (const float*)d_in[9];
  const float* bg1  = (const float*)d_in[10];
  const float* bu1  = (const float*)d_in[11];
  const float* z1   = (const float*)d_in[12];
  const float* nu1  = (const float*)d_in[13];
  const float* Wout = (const float*)d_in[14];
  const float* Bout = (const float*)d_in[15];
  float* out = (float*)d_out;
  _Float16* last0 = (_Float16*)d_ws;  // 16384*64 halfs = 2 MB scratch

  // Layer 0: 16384 sequences / 16 rows per wave-tile = 1024 single-wave WGs.
  fastgrnn_l0<<<1024, 32, 0, stream>>>(x, W0, U0, bg0, bu0, z0, nu0, last0);
  // Layer 1 + head: 256 rows / 16 = 16 single-wave WGs.
  fastgrnn_l1<<<16, 32, 0, stream>>>(last0, W1, U1, bg1, bu1, z1, nu1, Wout,
                                     Bout, out);
}
